// GCNTail_48936857370857
// MI455X (gfx1250) — compile-verified, matrix operand order checked by reading, not run on hardware
//
#include <hip/hip_runtime.h>
#include <hip/hip_bf16.h>

typedef __attribute__((ext_vector_type(16))) __bf16 v16bf;
typedef __attribute__((ext_vector_type(8)))  __bf16 v8bf;
typedef __attribute__((ext_vector_type(8)))  float  v8f;

// ---------------------------------------------------------------------------
// Small utility kernels
// ---------------------------------------------------------------------------
__global__ void fill_kernel(float* __restrict__ p, float v, int n) {
    int i = blockIdx.x * blockDim.x + threadIdx.x;
    if (i < n) p[i] = v;
}

__global__ void degree_kernel(const long long* __restrict__ dst,
                              float* __restrict__ deg, int E) {
    int e = blockIdx.x * blockDim.x + threadIdx.x;
    if (e < E) atomicAdd(&deg[dst[e]], 1.0f);
}

__global__ void rsqrt_kernel(float* __restrict__ d, int n) {
    int i = blockIdx.x * blockDim.x + threadIdx.x;
    if (i < n) d[i] = rsqrtf(d[i]);   // deg >= 1 always (self-loops)
}

// ---------------------------------------------------------------------------
// WMMA GEMM: Y[N,Fout] = X[N,K] @ W[K,Fout] (+ bias), bf16 inputs, f32 accum.
// 256 threads = 8 waves. Wave w -> colTile (w % colTiles), rowTile (w / colTiles).
// W staged in LDS column-major as bf16 so B fragments are contiguous.
// ---------------------------------------------------------------------------
__global__ void __launch_bounds__(256)
wmma_gemm_kernel(const float* __restrict__ X, const float* __restrict__ W,
                 const float* __restrict__ bias, float* __restrict__ Y,
                 int N, int K, int Fout, int FoutPad) {
    __shared__ __bf16 sW[128 * 64];   // max K*FoutPad = 128*64 = 16KB

    const int tid = threadIdx.x;
    // Stage W into LDS, column-major: sW[n*K + k], zero-pad cols >= Fout.
    const int total = K * FoutPad;
    for (int i = tid; i < total; i += 256) {
        int n = i / K;
        int k = i - n * K;
        sW[i] = (n < Fout) ? (__bf16)W[(size_t)k * Fout + n] : (__bf16)0.0f;
    }
    __syncthreads();

    const int wave     = tid >> 5;
    const int lane     = tid & 31;
    const int colTiles = FoutPad >> 4;
    const int rowWaves = 8 / colTiles;
    if (wave >= rowWaves * colTiles) return;   // uniform per-wave

    const int colTile = (wave % colTiles) << 4;
    const int rowTile = blockIdx.x * (rowWaves << 4) + ((wave / colTiles) << 4);

    const int hl    = lane & 15;
    const int khalf = (lane < 16) ? 0 : 8;     // A/B K-half per ISA layout
    const int row   = rowTile + hl;
    const int rowc  = (row < N) ? row : 0;     // clamp; garbage rows not stored
    const int colN  = colTile + hl;

    v8f acc = {};
    for (int kk = 0; kk < K; kk += 32) {
        // ---- A fragment: row `rowc`, K = kk+khalf+{0..7}, kk+khalf+16+{0..7}
        const float* xp = X + (size_t)rowc * K + kk + khalf;
        float4 a0 = *(const float4*)(xp);
        float4 a1 = *(const float4*)(xp + 4);
        float4 a2 = *(const float4*)(xp + 16);
        float4 a3 = *(const float4*)(xp + 20);
        v16bf a;
        a[0]  = (__bf16)a0.x; a[1]  = (__bf16)a0.y; a[2]  = (__bf16)a0.z; a[3]  = (__bf16)a0.w;
        a[4]  = (__bf16)a1.x; a[5]  = (__bf16)a1.y; a[6]  = (__bf16)a1.z; a[7]  = (__bf16)a1.w;
        a[8]  = (__bf16)a2.x; a[9]  = (__bf16)a2.y; a[10] = (__bf16)a2.z; a[11] = (__bf16)a2.w;
        a[12] = (__bf16)a3.x; a[13] = (__bf16)a3.y; a[14] = (__bf16)a3.z; a[15] = (__bf16)a3.w;

        // ---- B fragment: col `colN`, same K pattern; contiguous in LDS.
        const __bf16* wp = sW + (size_t)colN * K + kk + khalf;
        v8bf blo = *(const v8bf*)(wp);
        v8bf bhi = *(const v8bf*)(wp + 16);
        v16bf b = __builtin_shufflevector(blo, bhi,
                    0, 1, 2, 3, 4, 5, 6, 7, 8, 9, 10, 11, 12, 13, 14, 15);

        acc = __builtin_amdgcn_wmma_f32_16x16x32_bf16(
                  false, a, false, b, (short)0, acc, false, false);
    }

    // ---- Store C/D: VGPR r -> row rowTile + r (+8 for lanes 16..31), col colN.
    if (colN - colTile < 16 && (colTile + hl) < Fout) {
        const float bval  = bias ? bias[colTile + hl] : 0.0f;
        const int   rbase = rowTile + ((lane < 16) ? 0 : 8);
        #pragma unroll
        for (int r = 0; r < 8; ++r) {
            int rr = rbase + r;
            if (rr < N) Y[(size_t)rr * Fout + (colTile + hl)] = acc[r] + bval;
        }
    }
}

// ---------------------------------------------------------------------------
// Edge aggregation: out[dst] += h[src] * dinv[src]*dinv[dst]   (F = 64)
// 16 threads per edge, float4 gather + 4 f32 atomics (L2-resident).
// ---------------------------------------------------------------------------
__global__ void aggregate_kernel(const float* __restrict__ h,
                                 const long long* __restrict__ src,
                                 const long long* __restrict__ dst,
                                 const float* __restrict__ dinv,
                                 float* __restrict__ out, int E) {
    const int F = 64;
    int t  = blockIdx.x * blockDim.x + threadIdx.x;
    int e  = t >> 4;
    int f4 = (t & 15) << 2;
    if (e >= E) return;
    long long s = src[e];
    long long d = dst[e];
    float nrm = dinv[s] * dinv[d];
    float4 hv = *(const float4*)(h + (size_t)s * F + f4);
    float* op = out + (size_t)d * F + f4;
    atomicAdd(op + 0, hv.x * nrm);
    atomicAdd(op + 1, hv.y * nrm);
    atomicAdd(op + 2, hv.z * nrm);
    atomicAdd(op + 3, hv.w * nrm);
}

// ---------------------------------------------------------------------------
// Finish: agg[i,f] = [relu]( agg[i,f] + hpre[i,f]*dinv[i]^2 + bias[f] )
// (self-loop term folded in here; F = 64)
// ---------------------------------------------------------------------------
__global__ void finish_kernel(float* __restrict__ agg,
                              const float* __restrict__ hpre,
                              const float* __restrict__ dinv,
                              const float* __restrict__ bias,
                              int N, int doRelu) {
    int idx = blockIdx.x * blockDim.x + threadIdx.x;
    if (idx >= N * 64) return;
    int i = idx >> 6;
    int f = idx & 63;
    float di = dinv[i];
    float v  = agg[idx] + hpre[idx] * di * di + bias[f];
    if (doRelu) v = fmaxf(v, 0.0f);
    agg[idx] = v;
}

// ---------------------------------------------------------------------------
// Host-side launch
// ---------------------------------------------------------------------------
static inline void launch_gemm(const float* X, const float* W, const float* bias,
                               float* Y, int N, int K, int Fout, hipStream_t stream) {
    int FoutPad      = (Fout + 15) & ~15;
    int colTiles     = FoutPad >> 4;
    int rowWaves     = 8 / colTiles;
    int rowsPerBlock = rowWaves << 4;
    int grid         = (N + rowsPerBlock - 1) / rowsPerBlock;
    wmma_gemm_kernel<<<grid, 256, 0, stream>>>(X, W, bias, Y, N, K, Fout, FoutPad);
}

extern "C" void kernel_launch(void* const* d_in, const int* in_sizes, int n_in,
                              void* d_out, int out_size, void* d_ws, size_t ws_size,
                              hipStream_t stream) {
    const float*     x  = (const float*)d_in[0];
    const long long* ei = (const long long*)d_in[1];   // int64 [2,E]
    const float*     W1 = (const float*)d_in[2];
    const float*     b1 = (const float*)d_in[3];
    const float*     W2 = (const float*)d_in[4];
    const float*     b2 = (const float*)d_in[5];
    const float*     Wc = (const float*)d_in[6];
    const float*     bc = (const float*)d_in[7];
    float*           out = (float*)d_out;

    const int Fin = 128, H = 64;
    const int N = in_sizes[0] / Fin;       // 100000
    const int E = in_sizes[1] / 2;         // 1000000
    const int C = out_size / N;            // 40

    const long long* srcI = ei;
    const long long* dstI = ei + E;

    // Workspace layout (256B aligned): dinv[N] | bufA[N*H] | bufB[N*H]
    char*  ws   = (char*)d_ws;
    size_t off0 = ((size_t)N * sizeof(float) + 255) & ~(size_t)255;
    float* dinv = (float*)ws;
    float* bufA = (float*)(ws + off0);
    float* bufB = bufA + (size_t)N * H;

    const int TB = 256;

    // Degrees (self-loop => init 1.0), then dinv = rsqrt(deg).
    fill_kernel<<<(N + TB - 1) / TB, TB, 0, stream>>>(dinv, 1.0f, N);
    degree_kernel<<<(E + TB - 1) / TB, TB, 0, stream>>>(dstI, dinv, E);
    rsqrt_kernel<<<(N + TB - 1) / TB, TB, 0, stream>>>(dinv, N);

    // ---- Layer 1: bufA = x @ W1 ; aggregate into bufB ; finish(+b1, relu)
    launch_gemm(x, W1, nullptr, bufA, N, Fin, H, stream);
    hipMemsetAsync(bufB, 0, (size_t)N * H * sizeof(float), stream);
    aggregate_kernel<<<((E * 16) + TB - 1) / TB, TB, 0, stream>>>(
        bufA, srcI, dstI, dinv, bufB, E);
    finish_kernel<<<((N * H) + TB - 1) / TB, TB, 0, stream>>>(
        bufB, bufA, dinv, b1, N, 1);

    // ---- Layer 2: bufA = bufB @ W2 ; aggregate into bufB ; finish(+b2, relu)
    launch_gemm(bufB, W2, nullptr, bufA, N, H, H, stream);
    hipMemsetAsync(bufB, 0, (size_t)N * H * sizeof(float), stream);
    aggregate_kernel<<<((E * 16) + TB - 1) / TB, TB, 0, stream>>>(
        bufA, srcI, dstI, dinv, bufB, E);
    finish_kernel<<<((N * H) + TB - 1) / TB, TB, 0, stream>>>(
        bufB, bufA, dinv, b2, N, 1);

    // ---- Classifier: out = bufB @ Wc + bc   (Fout = 40, padded to 48)
    launch_gemm(bufB, Wc, bc, out, N, H, C, stream);
}
